// MMDLoss_38414187495592
// MI455X (gfx1250) — matrix operand order repaired and checked by reading, not compiled, not run on hardware
//
#include <hip/hip_runtime.h>
#include <hip/hip_bf16.h>

// MMD loss on MI455X (gfx1250, wave32, WMMA).
// source: 4096x512 f32, target: 4096x512 f32, output: 1 f32 scalar.
//
// Pipeline:
//  1) pack:   fp32 -> bf16 hi/lo split (x ~= hi + lo), per-row squared norms
//  2) bw:     bandwidth via closed form  sum(l2) = 2N*sum(sq) - 2*||colsum||^2
//  3) gemm:   G = A*A^T with 3-term bf16 WMMA (hi*hi + hi*lo + lo*hi),
//             fused epilogue: l2 -> sum of 5 Gaussians -> quadrant-signed sum
//  4) reduce: deterministic tree sum of 4096 block partials, /(4096^2)
//
// Workspace layout (needs ~16.1 MB):
//   [0,                8M)  a_hi  : 8192*512 bf16
//   [8M,              16M)  a_lo  : 8192*512 bf16
//   [16M,      16M+32KB)    sq    : 8192 f32
//   nib   : 8 f32 (5 used: -1/(bw*2^t))
//   partial: 4096 f32 block partials

#define NROWS 8192
#define BROWS 4096
#define DIMK  512

typedef __attribute__((ext_vector_type(16))) __bf16 v16bf;
typedef __attribute__((ext_vector_type(8)))  float  v8f;
typedef __attribute__((ext_vector_type(4)))  unsigned int u32x4;

union Frag {
    v16bf v;
    u32x4 u[2];
};

// ---------------------------------------------------------------- pack kernel
__global__ __launch_bounds__(256) void mmd_pack(
    const float* __restrict__ src, const float* __restrict__ tgt,
    __bf16* __restrict__ ahi, __bf16* __restrict__ alo, float* __restrict__ sq)
{
    const int row = blockIdx.x;                 // 0..8191
    const float* a = (row < BROWS) ? (src + (size_t)row * DIMK)
                                   : (tgt + (size_t)(row - BROWS) * DIMK);
    __shared__ float red[256];
    float s = 0.f;
    for (int t = threadIdx.x; t < DIMK; t += 256) {
        float x  = a[t];
        __bf16 h = (__bf16)x;                   // RNE truncate to bf16
        float  l = x - (float)h;                // residual fits in bf16 exactly enough
        ahi[(size_t)row * DIMK + t] = h;
        alo[(size_t)row * DIMK + t] = (__bf16)l;
        s += x * x;
    }
    red[threadIdx.x] = s;
    __syncthreads();
    for (int o = 128; o > 0; o >>= 1) {
        if (threadIdx.x < o) red[threadIdx.x] += red[threadIdx.x + o];
        __syncthreads();
    }
    if (threadIdx.x == 0) sq[row] = red[0];
}

// ------------------------------------------------------------ bandwidth kernel
// sum_ij l2_ij = 2*N*sum_i sq_i - 2*||sum_i a_i||^2   (exact identity)
__global__ __launch_bounds__(512) void mmd_bw(
    const float* __restrict__ src, const float* __restrict__ tgt,
    const float* __restrict__ sq, float* __restrict__ nib)
{
    __shared__ float red[512];
    __shared__ float ssq_s;
    const int d = threadIdx.x;                  // 0..511 = feature dim
    float cs = 0.f;
    for (int r = 0; r < BROWS; ++r) cs += src[(size_t)r * DIMK + d];
    for (int r = 0; r < BROWS; ++r) cs += tgt[(size_t)r * DIMK + d];
    // reduce ||colsum||^2
    red[d] = cs * cs;
    __syncthreads();
    for (int o = 256; o > 0; o >>= 1) {
        if (d < o) red[d] += red[d + o];
        __syncthreads();
    }
    if (d == 0) ssq_s = red[0];
    __syncthreads();
    // reduce sum(sq)
    float sp = 0.f;
    for (int r = d; r < NROWS; r += 512) sp += sq[r];
    red[d] = sp;
    __syncthreads();
    for (int o = 256; o > 0; o >>= 1) {
        if (d < o) red[d] += red[d + o];
        __syncthreads();
    }
    if (d == 0) {
        float sum_l2 = 2.0f * (float)NROWS * red[0] - 2.0f * ssq_s;
        float bw = sum_l2 / (float)((double)NROWS * (double)NROWS - (double)NROWS);
        float m = 1.0f;
        for (int t = 0; t < 5; ++t) { nib[t] = -1.0f / (bw * m); m *= 2.0f; }
    }
}

// --------------------------------------------------------------- main WMMA GEMM
// Block = 256 threads = 8 waves; block tile 128x128.
// Wave (wm in 0..1, wn in 0..3) computes a 64x32 region = 4x2 sub-tiles of 16x16.
// Per K-chunk of 32: 3 bf16 WMMAs per sub-tile (hi*hi + hi*lo + lo*hi).
__global__ __launch_bounds__(256) void mmd_gemm(
    const __bf16* __restrict__ ahi, const __bf16* __restrict__ alo,
    const float* __restrict__ sq, const float* __restrict__ nib,
    float* __restrict__ partial)
{
    const int tid    = threadIdx.x;
    const int lane   = tid & 31;
    const int wave   = tid >> 5;        // 0..7
    const int waveM  = wave >> 2;       // 0..1
    const int waveN  = wave & 3;        // 0..3
    const int lane16 = lane & 15;
    const int hsel   = (lane >> 4) & 1; // which half-wave
    const int bM     = (int)blockIdx.x * 128;
    const int bN     = (int)blockIdx.y * 128;

    __shared__ float sqM[128];
    __shared__ float sqN[128];
    __shared__ float nibs[5];
    __shared__ float red[256];

    if (tid < 128)      sqM[tid]       = sq[bM + tid];
    else                sqN[tid - 128] = sq[bN + (tid - 128)];
    if (tid < 5)        nibs[tid]      = nib[tid];
    __syncthreads();

    v8f acc[4][2];
#pragma unroll
    for (int mt = 0; mt < 4; ++mt)
#pragma unroll
        for (int nt = 0; nt < 2; ++nt) acc[mt][nt] = {};

    // Per-lane base rows (ISA 16-bit A 16x32 layout: lanes 0-15 / 16-31 share M)
    size_t aBase[4], bBase[2];
#pragma unroll
    for (int mt = 0; mt < 4; ++mt)
        aBase[mt] = (size_t)(bM + waveM * 64 + mt * 16 + lane16) * DIMK;
#pragma unroll
    for (int nt = 0; nt < 2; ++nt)
        bBase[nt] = (size_t)(bN + waveN * 32 + nt * 16 + lane16) * DIMK;

    for (int kt = 0; kt < DIMK / 32; ++kt) {
        const int k0 = kt * 32;
        Frag ah[4], al[4], bh[2], bl[2];
        // A fragment (16x32 bf16): elems 0..7 = K k0+hsel*8.., elems 8..15 = K k0+16+hsel*8..
#pragma unroll
        for (int mt = 0; mt < 4; ++mt) {
            size_t o0 = aBase[mt] + (size_t)(k0 + hsel * 8);
            ah[mt].u[0] = *(const u32x4*)(ahi + o0);
            ah[mt].u[1] = *(const u32x4*)(ahi + o0 + 16);
            al[mt].u[0] = *(const u32x4*)(alo + o0);
            al[mt].u[1] = *(const u32x4*)(alo + o0 + 16);
        }
        // B fragment (32x16 bf16, column-major): lane holds col n=lane16,
        // elems 0..7 = K k0+hsel*8.., elems 8..15 = K k0+16+hsel*8..
#pragma unroll
        for (int nt = 0; nt < 2; ++nt) {
            size_t o0 = bBase[nt] + (size_t)(k0 + hsel * 8);
            bh[nt].u[0] = *(const u32x4*)(ahi + o0);
            bh[nt].u[1] = *(const u32x4*)(ahi + o0 + 16);
            bl[nt].u[0] = *(const u32x4*)(alo + o0);
            bl[nt].u[1] = *(const u32x4*)(alo + o0 + 16);
        }
        // 3-term split product: full-precision-class A*A^T
#pragma unroll
        for (int mt = 0; mt < 4; ++mt)
#pragma unroll
            for (int nt = 0; nt < 2; ++nt) {
                acc[mt][nt] = __builtin_amdgcn_wmma_f32_16x16x32_bf16(
                    false, ah[mt].v, false, bh[nt].v, (short)0, acc[mt][nt], false, false);
                acc[mt][nt] = __builtin_amdgcn_wmma_f32_16x16x32_bf16(
                    false, ah[mt].v, false, bl[nt].v, (short)0, acc[mt][nt], false, false);
                acc[mt][nt] = __builtin_amdgcn_wmma_f32_16x16x32_bf16(
                    false, al[mt].v, false, bh[nt].v, (short)0, acc[mt][nt], false, false);
            }
    }

    // Epilogue: l2 -> 5 Gaussian kernels -> quadrant sign -> local sum.
    // C/D layout: VGPR r, lanes 0-15: M=r, N=lane; lanes 16-31: M=r+8, N=lane-16
    float local = 0.f;
    const float n0 = nibs[0], n1 = nibs[1], n2 = nibs[2], n3 = nibs[3], n4 = nibs[4];
#pragma unroll
    for (int mt = 0; mt < 4; ++mt)
#pragma unroll
        for (int nt = 0; nt < 2; ++nt) {
            const int colL = waveN * 32 + nt * 16 + lane16;
            const int j    = bN + colL;
            const float sqj = sqN[colL];
#pragma unroll
            for (int r = 0; r < 8; ++r) {
                const int rowL = waveM * 64 + mt * 16 + r + hsel * 8;
                const int i    = bM + rowL;
                float l2 = sqM[rowL] + sqj - 2.0f * acc[mt][nt][r];
                float kv = __expf(l2 * n0) + __expf(l2 * n1) + __expf(l2 * n2)
                         + __expf(l2 * n3) + __expf(l2 * n4);
                local += ((i < BROWS) == (j < BROWS)) ? kv : -kv;
            }
        }

    red[tid] = local;
    __syncthreads();
    for (int o = 128; o > 0; o >>= 1) {
        if (tid < o) red[tid] += red[tid + o];
        __syncthreads();
    }
    if (tid == 0) partial[(int)blockIdx.y * 64 + (int)blockIdx.x] = red[0];
}

// ------------------------------------------------------------- final reduction
__global__ __launch_bounds__(256) void mmd_reduce(
    const float* __restrict__ partial, float* __restrict__ out)
{
    __shared__ float red[256];
    float s = 0.f;
    for (int t = threadIdx.x; t < 4096; t += 256) s += partial[t];
    red[threadIdx.x] = s;
    __syncthreads();
    for (int o = 128; o > 0; o >>= 1) {
        if (threadIdx.x < o) red[threadIdx.x] += red[threadIdx.x + o];
        __syncthreads();
    }
    if (threadIdx.x == 0) out[0] = red[0] / ((float)BROWS * (float)BROWS);
}

// --------------------------------------------------------------------- launch
extern "C" void kernel_launch(void* const* d_in, const int* in_sizes, int n_in,
                              void* d_out, int out_size, void* d_ws, size_t ws_size,
                              hipStream_t stream) {
    const float* src = (const float*)d_in[0];
    const float* tgt = (const float*)d_in[1];

    char* ws = (char*)d_ws;
    const size_t nElems = (size_t)NROWS * DIMK;           // 4 Mi elements
    __bf16* ahi = (__bf16*)ws;                            // 8 MB
    __bf16* alo = (__bf16*)(ws + nElems * sizeof(__bf16));// 8 MB
    float*  sq  = (float*)(ws + nElems * sizeof(__bf16) * 2);
    float*  nib = sq + NROWS;
    float*  partial = nib + 8;                            // 4096 f32

    mmd_pack<<<NROWS, 256, 0, stream>>>(src, tgt, ahi, alo, sq);
    mmd_bw<<<1, 512, 0, stream>>>(src, tgt, sq, nib);
    dim3 grid(64, 64);
    mmd_gemm<<<grid, 256, 0, stream>>>(ahi, alo, sq, nib, partial);
    mmd_reduce<<<1, 256, 0, stream>>>(partial, (float*)d_out);
}
